// StdHead_45938970198294
// MI455X (gfx1250) — matrix operand order repaired and checked by reading, not compiled, not run on hardware
//
#include <hip/hip_runtime.h>
#include <hip/hip_bf16.h>

#define B_   8
#define CIN  1152
#define NPTS 4096
#define CH   128
#define KNN  32
#define TM   64

#define XS_STRIDE 40    // ushorts; 80B rows: 16B aligned, conflict-free fragment reads
#define KS_STRIDE 136   // ushorts; 272B rows: 16B aligned, conflict-free fragment reads
#define FB_STRIDE 132   // floats;  528B rows: 16B aligned

typedef __attribute__((ext_vector_type(16))) __bf16 v16bf;
typedef __attribute__((ext_vector_type(2)))  __bf16 v2bf;
typedef __attribute__((ext_vector_type(8)))  float  v8f;
typedef __attribute__((ext_vector_type(4)))  int    v4i;
typedef unsigned short ushort_t;
typedef unsigned int   uint_t;

#if defined(__has_builtin)
#if __has_builtin(__builtin_amdgcn_cvt_pk_bf16_f32)
#define HAVE_CVT_PK_BF16 1
#endif
#if __has_builtin(__builtin_amdgcn_global_load_async_to_lds_b128)
#define HAVE_ASYNC_LDS 1
#endif
#endif

typedef __attribute__((address_space(1))) v4i* gptr_v4i;
typedef __attribute__((address_space(3))) v4i* lptr_v4i;

// ---------- small helpers ----------
__device__ __forceinline__ ushort_t f2bf_us(float f) {
  uint_t u = __float_as_uint(f);
  uint_t r = u + 0x7FFFu + ((u >> 16) & 1u);   // round-to-nearest-even
  return (ushort_t)(r >> 16);
}
__device__ __forceinline__ uint_t pack_bf2(float lo, float hi) {
#if HAVE_CVT_PK_BF16
  v2bf p = __builtin_amdgcn_cvt_pk_bf16_f32(lo, hi);
  return __builtin_bit_cast(uint_t, p);
#else
  return (uint_t)f2bf_us(lo) | ((uint_t)f2bf_us(hi) << 16);
#endif
}
__device__ __forceinline__ __bf16 us2bf(ushort_t h) {
  return __builtin_bit_cast(__bf16, h);
}
__device__ __forceinline__ v8f zero8() {
  v8f z;
#pragma unroll
  for (int i = 0; i < 8; ++i) z[i] = 0.f;
  return z;
}
__device__ __forceinline__ v8f wmma_bf16(v16bf a, v16bf b, v8f c) {
  return __builtin_amdgcn_wmma_f32_16x16x32_bf16(false, a, false, b, (short)0, c,
                                                 false, false);
}

// Copy 16B global -> LDS. Uses gfx1250 async DMA path when available.
__device__ __forceinline__ void stage16(ushort_t* lds_dst, const ushort_t* gsrc) {
#if HAVE_ASYNC_LDS
  __builtin_amdgcn_global_load_async_to_lds_b128(
      (gptr_v4i)(ushort_t*)gsrc, (lptr_v4i)lds_dst, 0, 0);
#else
  *(uint4*)lds_dst = *(const uint4*)gsrc;
#endif
}
__device__ __forceinline__ void stage_drain() {
#if HAVE_ASYNC_LDS
  asm volatile("s_wait_asynccnt 0x0" ::: "memory");
#endif
}

// A-matrix fragment: 16x32 (MxK) bf16 tile, LDS row-major [m][k], k contiguous.
// lanes 0-15: M=lane, K={0..7,16..23}; lanes 16-31: M=lane-16, K={8..15,24..31}
__device__ __forceinline__ v16bf load_a(const ushort_t* s, int stride, int lane) {
  int m = lane & 15;
  int kb = (lane < 16) ? 0 : 8;
  v16bf a;
#pragma unroll
  for (int e = 0; e < 16; ++e) {
    int k = kb + e + ((e < 8) ? 0 : 8);
    a[e] = us2bf(s[m * stride + k]);
  }
  return a;
}
// B-matrix fragment from an n-major tile [n][k] (k contiguous): B[k][n] = tile[n][k].
__device__ __forceinline__ v16bf load_bt(const ushort_t* s, int stride, int lane) {
  int n = lane & 15;
  int kb = (lane < 16) ? 0 : 8;
  v16bf b;
#pragma unroll
  for (int e = 0; e < 16; ++e) b[e] = us2bf(s[n * stride + kb + e + ((e < 8) ? 0 : 8)]);
  return b;
}

// ---------- prep kernels ----------
// W1 stays [o][c] row-major (exactly the n-major B-tile layout we want).
__global__ __launch_bounds__(256) void k_convert_w1(const float* __restrict__ W1,
                                                    ushort_t* __restrict__ w1c) {
  int e = blockIdx.x * 256 + threadIdx.x;   // e < CH*CIN/2
  float lo = W1[2 * e], hi = W1[2 * e + 1];
  ((uint_t*)w1c)[e] = pack_bf2(lo, hi);
}

// A_t[j][i] = sum_o Wq[o][i] * Wk[o][j]   (transposed A = Wq^T Wk, n-major for B tiles)
__global__ __launch_bounds__(128) void k_compute_A(const float* __restrict__ Wq,
                                                   const float* __restrict__ Wk,
                                                   ushort_t* __restrict__ A_t) {
  int i = blockIdx.x, j = threadIdx.x;
  float acc = 0.f;
  for (int o = 0; o < CH; ++o) acc += Wq[o * CH + i] * Wk[o * CH + j];
  A_t[(size_t)j * CH + i] = f2bf_us(acc);
}

// ---------- GEMM1 + BN + LeakyReLU -> f(bf16), sq ----------
__global__ __launch_bounds__(256) void k_gemm1(
    const float* __restrict__ x, const ushort_t* __restrict__ w1c,
    const float* __restrict__ gamma, const float* __restrict__ beta,
    const float* __restrict__ mean, const float* __restrict__ var,
    ushort_t* __restrict__ f_bf, float* __restrict__ sq) {
  __shared__ ushort_t Xs[TM * XS_STRIDE];    // A tile: 64(m) x 32(k)
  __shared__ ushort_t Ws[CH * XS_STRIDE];    // B tile, n-major: 128(o) x 32(k)
  __shared__ float    fbuf[TM * FB_STRIDE];

  const int tid = threadIdx.x, wave = tid >> 5, lane = tid & 31;
  const int b = blockIdx.y;
  const int n0 = blockIdx.x * TM;
  const int mt = wave >> 1, nh = wave & 1;

  // x staging map: thread owns row m, contiguous k octet k0..k0+7
  const int xm = tid & 63, xk0 = (tid >> 6) * 8;
  const float* xp = x + ((size_t)b * CIN + xk0) * NPTS + n0 + xm;
  // W staging map: 2 rounds of (row o, octet ch)
  const int wo = tid >> 2, wch = tid & 3;

  v8f acc[4];
#pragma unroll
  for (int j = 0; j < 4; ++j) acc[j] = zero8();

  for (int c0 = 0; c0 < CIN; c0 += 32) {
    {  // A tile: 8 f32 loads (base + imm offsets) -> packed b128 LDS store
      uint4 pk;
      pk.x = pack_bf2(xp[0 * NPTS], xp[1 * NPTS]);
      pk.y = pack_bf2(xp[2 * NPTS], xp[3 * NPTS]);
      pk.z = pack_bf2(xp[4 * NPTS], xp[5 * NPTS]);
      pk.w = pack_bf2(xp[6 * NPTS], xp[7 * NPTS]);
      *(uint4*)&Xs[xm * XS_STRIDE + xk0] = pk;
    }
#pragma unroll
    for (int i = 0; i < 2; ++i) {  // B tile: b128 copies
      int o = wo + i * 64;
      *(uint4*)&Ws[o * XS_STRIDE + wch * 8] =
          *(const uint4*)&w1c[(size_t)o * CIN + c0 + wch * 8];
    }
    if (c0 + 32 < CIN)                       // hint next A tile (global_prefetch_b8)
      __builtin_prefetch(xp + (size_t)32 * NPTS, 0, 0);
    xp += (size_t)32 * NPTS;
    __syncthreads();
    v16bf af  = load_a(&Xs[mt * 16 * XS_STRIDE], XS_STRIDE, lane);
    v16bf bf0 = load_bt(&Ws[(nh * 4 + 0) * 16 * XS_STRIDE], XS_STRIDE, lane);
    v16bf bf1 = load_bt(&Ws[(nh * 4 + 1) * 16 * XS_STRIDE], XS_STRIDE, lane);
    v16bf bf2 = load_bt(&Ws[(nh * 4 + 2) * 16 * XS_STRIDE], XS_STRIDE, lane);
    v16bf bf3 = load_bt(&Ws[(nh * 4 + 3) * 16 * XS_STRIDE], XS_STRIDE, lane);
    acc[0] = wmma_bf16(af, bf0, acc[0]);
    acc[1] = wmma_bf16(af, bf1, acc[1]);
    acc[2] = wmma_bf16(af, bf2, acc[2]);
    acc[3] = wmma_bf16(af, bf3, acc[3]);
    __syncthreads();
  }

#pragma unroll
  for (int jj = 0; jj < 4; ++jj) {
    int col = (nh * 4 + jj) * 16 + (lane & 15);
    float sc = gamma[col] * rsqrtf(var[col] + 1e-5f);
    float sh = beta[col] - mean[col] * sc;
#pragma unroll
    for (int r = 0; r < 8; ++r) {
      int row = mt * 16 + r + ((lane < 16) ? 0 : 8);
      float h = acc[jj][r] * sc + sh;
      h = (h >= 0.f) ? h : 0.2f * h;       // LeakyReLU(0.2)
      fbuf[row * FB_STRIDE + col] = h;
    }
  }
  __syncthreads();
#pragma unroll
  for (int i = 0; i < 4; ++i) {            // pack 8 cols -> one b128 store
    int e = tid + i * 256;
    int row = e >> 4, ch = e & 15;
    const float* fp = &fbuf[row * FB_STRIDE + ch * 8];
    uint4 pk;
    pk.x = pack_bf2(fp[0], fp[1]);
    pk.y = pack_bf2(fp[2], fp[3]);
    pk.z = pack_bf2(fp[4], fp[5]);
    pk.w = pack_bf2(fp[6], fp[7]);
    *(uint4*)&f_bf[((size_t)b * NPTS + n0 + row) * CH + ch * 8] = pk;
  }
  if (tid < TM) {
    const float4* fp = (const float4*)&fbuf[tid * FB_STRIDE];
    float s = 0.f;
#pragma unroll
    for (int i = 0; i < 32; ++i) {
      float4 v = fp[i];
      s += v.x * v.x + v.y * v.y + v.z * v.z + v.w * v.w;
    }
    sq[(size_t)b * NPTS + n0 + tid] = s;
  }
}

// ---------- g = f @ A  (f32 out) ----------
__global__ __launch_bounds__(256) void k_gemm_g(const ushort_t* __restrict__ f_bf,
                                                const ushort_t* __restrict__ A_t,
                                                float* __restrict__ g) {
  __shared__ ushort_t Xs[TM * XS_STRIDE];
  __shared__ ushort_t Ws[CH * XS_STRIDE];
  const int tid = threadIdx.x, wave = tid >> 5, lane = tid & 31;
  const int b = blockIdx.y;
  const int n0 = blockIdx.x * TM;
  const int mt = wave >> 1, nh = wave & 1;
  const int xm = tid >> 2, xch = tid & 3;    // A-tile map (64 x 4 octets)
  const int wo = tid >> 2, wch = tid & 3;    // B-tile map

  v8f acc[4];
#pragma unroll
  for (int j = 0; j < 4; ++j) acc[j] = zero8();

#pragma unroll
  for (int c0 = 0; c0 < CH; c0 += 32) {
    *(uint4*)&Xs[xm * XS_STRIDE + xch * 8] =
        *(const uint4*)&f_bf[((size_t)b * NPTS + n0 + xm) * CH + c0 + xch * 8];
#pragma unroll
    for (int i = 0; i < 2; ++i) {
      int o = wo + i * 64;
      *(uint4*)&Ws[o * XS_STRIDE + wch * 8] =
          *(const uint4*)&A_t[(size_t)o * CH + c0 + wch * 8];
    }
    __syncthreads();
    v16bf af  = load_a(&Xs[mt * 16 * XS_STRIDE], XS_STRIDE, lane);
    v16bf bf0 = load_bt(&Ws[(nh * 4 + 0) * 16 * XS_STRIDE], XS_STRIDE, lane);
    v16bf bf1 = load_bt(&Ws[(nh * 4 + 1) * 16 * XS_STRIDE], XS_STRIDE, lane);
    v16bf bf2 = load_bt(&Ws[(nh * 4 + 2) * 16 * XS_STRIDE], XS_STRIDE, lane);
    v16bf bf3 = load_bt(&Ws[(nh * 4 + 3) * 16 * XS_STRIDE], XS_STRIDE, lane);
    acc[0] = wmma_bf16(af, bf0, acc[0]);
    acc[1] = wmma_bf16(af, bf1, acc[1]);
    acc[2] = wmma_bf16(af, bf2, acc[2]);
    acc[3] = wmma_bf16(af, bf3, acc[3]);
    __syncthreads();
  }
#pragma unroll
  for (int jj = 0; jj < 4; ++jj) {
    int col = (nh * 4 + jj) * 16 + (lane & 15);
#pragma unroll
    for (int r = 0; r < 8; ++r) {
      int row = mt * 16 + r + ((lane < 16) ? 0 : 8);
      g[((size_t)b * NPTS + n0 + row) * CH + col] = acc[jj][r];
    }
  }
}

// ---------- KNN: dist = sq_n + sq_m - 2 f_n.f_m, running top-32 ----------
__global__ __launch_bounds__(256) void k_knn(const ushort_t* __restrict__ f_bf,
                                             const float* __restrict__ sq,
                                             int* __restrict__ idx) {
  __shared__ ushort_t As[TM * KS_STRIDE];
  __shared__ ushort_t Bs[TM * KS_STRIDE];
  __shared__ float    dbuf[TM * 65];
  const int tid = threadIdx.x, wave = tid >> 5, lane = tid & 31;
  const int b = blockIdx.y;
  const int n0 = blockIdx.x * TM;
  const int mt = wave >> 1;              // shared M-tile for this wave's 2 outputs
  const int ntA = (wave & 1) * 2;        // first of 2 N-tiles

#pragma unroll
  for (int i = 0; i < 4; ++i) {          // stage 64 row features once
    int e = tid + i * 256;
    int m = e >> 4, ch = e & 15;
    stage16(&As[m * KS_STRIDE + ch * 8],
            &f_bf[((size_t)b * NPTS + n0 + m) * CH + ch * 8]);
  }
  stage_drain();

  float bestd[KNN];
  int   besti[KNN];
#pragma unroll
  for (int s0 = 0; s0 < KNN; ++s0) { bestd[s0] = 3.4e38f; besti[s0] = 0; }
  float worst = 3.4e38f;
  int   wslot = 0;

  for (int m0 = 0; m0 < NPTS; m0 += TM) {
    __syncthreads();
#pragma unroll
    for (int i = 0; i < 4; ++i) {
      int e = tid + i * 256;
      int m = e >> 4, ch = e & 15;
      stage16(&Bs[m * KS_STRIDE + ch * 8],
              &f_bf[((size_t)b * NPTS + m0 + m) * CH + ch * 8]);
      if (m0 + TM < NPTS)
        __builtin_prefetch(&f_bf[((size_t)b * NPTS + m0 + TM + m) * CH + ch * 8], 0, 0);
    }
    stage_drain();
    __syncthreads();

    v8f acc0 = zero8(), acc1 = zero8();
#pragma unroll
    for (int c0 = 0; c0 < CH; c0 += 32) {
      v16bf af  = load_a(&As[(mt * 16) * KS_STRIDE + c0], KS_STRIDE, lane);
      v16bf bf0 = load_bt(&Bs[(ntA * 16) * KS_STRIDE + c0], KS_STRIDE, lane);
      v16bf bf1 = load_bt(&Bs[((ntA + 1) * 16) * KS_STRIDE + c0], KS_STRIDE, lane);
      acc0 = wmma_bf16(af, bf0, acc0);
      acc1 = wmma_bf16(af, bf1, acc1);
    }
#pragma unroll
    for (int p = 0; p < 2; ++p) {
      int nt = ntA + p;
      int col_l = nt * 16 + (lane & 15);
      float sqm = sq[(size_t)b * NPTS + m0 + col_l];
      v8f a = p ? acc1 : acc0;
#pragma unroll
      for (int r = 0; r < 8; ++r) {
        int row_l = mt * 16 + r + ((lane < 16) ? 0 : 8);
        float sqn = sq[(size_t)b * NPTS + n0 + row_l];
        dbuf[row_l * 65 + col_l] = sqn + sqm - 2.f * a[r];
      }
    }
    __syncthreads();
    if (tid < TM) {
      for (int c = 0; c < TM; ++c) {
        float d = dbuf[tid * 65 + c];
        if (d < worst) {
#pragma unroll
          for (int s0 = 0; s0 < KNN; ++s0)
            if (s0 == wslot) { bestd[s0] = d; besti[s0] = m0 + c; }
          float w = bestd[0];
          int sl = 0;
#pragma unroll
          for (int s0 = 1; s0 < KNN; ++s0)
            if (bestd[s0] > w) { w = bestd[s0]; sl = s0; }
          worst = w;
          wslot = sl;
        }
      }
    }
  }
  if (tid < TM) {
#pragma unroll
    for (int k = 0; k < KNN; ++k)
      idx[((size_t)b * NPTS + n0 + tid) * KNN + k] = besti[k];
  }
}

// ---------- attention: one wave per point, lane = neighbor ----------
__global__ __launch_bounds__(256) void k_attn(const ushort_t* __restrict__ f_bf,
                                              const float* __restrict__ g,
                                              const int* __restrict__ idx,
                                              float* __restrict__ s_out) {
  __shared__ float gsh[8 * 128];
  const int tid = threadIdx.x, wave = tid >> 5, lane = tid & 31;
  const int pt = blockIdx.x * 8 + wave;    // flat b*N + n
  const int b = pt >> 12;

#pragma unroll
  for (int i = 0; i < 4; ++i)
    gsh[wave * 128 + lane + i * 32] = g[(size_t)pt * CH + lane + i * 32];
  __syncthreads();

  int nb = idx[(size_t)pt * KNN + lane];
  const uint4* fn4 = (const uint4*)(f_bf + ((size_t)b * NPTS + nb) * CH);
  float e = 0.f;
#pragma unroll
  for (int i = 0; i < 16; ++i) {           // 128 bf16 dims, 8 per b128 load
    uint4 u = fn4[i];
    const float* gp = &gsh[wave * 128 + 8 * i];
    e += __uint_as_float(u.x << 16) * gp[0];
    e += __uint_as_float(u.x & 0xFFFF0000u) * gp[1];
    e += __uint_as_float(u.y << 16) * gp[2];
    e += __uint_as_float(u.y & 0xFFFF0000u) * gp[3];
    e += __uint_as_float(u.z << 16) * gp[4];
    e += __uint_as_float(u.z & 0xFFFF0000u) * gp[5];
    e += __uint_as_float(u.w << 16) * gp[6];
    e += __uint_as_float(u.w & 0xFFFF0000u) * gp[7];
  }
  e *= 0.08838834764831845f;               // 1/sqrt(128)

  float m = e;
#pragma unroll
  for (int off = 16; off; off >>= 1) m = fmaxf(m, __shfl_xor(m, off));
  float ex = __expf(e - m);
  float ssum = ex;
#pragma unroll
  for (int off = 16; off; off >>= 1) ssum += __shfl_xor(ssum, off);
  float a = ex / ssum;
  float s2 = a * a;
#pragma unroll
  for (int off = 16; off; off >>= 1) s2 += __shfl_xor(s2, off);
  if (lane == 0) {
    // softmax sums to 1 => mean over K is exactly 1/32
    float var = s2 * (1.f / 32.f) - (1.f / 1024.f);
    s_out[pt] = sqrtf(fmaxf(var, 0.f));
  }
}

// ---------- per-batch standardize + modified sigmoid ----------
__global__ __launch_bounds__(256) void k_final(const float* __restrict__ s,
                                               float* __restrict__ out) {
  __shared__ float r1[256], r2[256];
  const int b = blockIdx.x, tid = threadIdx.x;
  float sum = 0.f, sum2 = 0.f;
  for (int i = tid; i < NPTS; i += 256) {
    float v = s[(size_t)b * NPTS + i];
    sum += v;
    sum2 += v * v;
  }
  r1[tid] = sum;
  r2[tid] = sum2;
  __syncthreads();
  for (int st = 128; st; st >>= 1) {
    if (tid < st) { r1[tid] += r1[tid + st]; r2[tid] += r2[tid + st]; }
    __syncthreads();
  }
  float mean = r1[0] * (1.f / NPTS);
  float varu = (r2[0] - (float)NPTS * mean * mean) * (1.f / (NPTS - 1));
  float inv = rsqrtf(fmaxf(varu, 1e-30f));
  for (int i = tid; i < NPTS; i += 256) {
    float z = (s[(size_t)b * NPTS + i] - mean) * inv;
    out[(size_t)b * NPTS + i] = 1.f / (1.f + __expf(-z * 2.9957322735539909f));
  }
}

// ---------- launch ----------
extern "C" void kernel_launch(void* const* d_in, const int* in_sizes, int n_in,
                              void* d_out, int out_size, void* d_ws, size_t ws_size,
                              hipStream_t stream) {
  (void)in_sizes; (void)n_in; (void)out_size; (void)ws_size;
  const float* x     = (const float*)d_in[0];
  const float* W1    = (const float*)d_in[1];
  const float* gamma = (const float*)d_in[2];
  const float* beta  = (const float*)d_in[3];
  const float* mean  = (const float*)d_in[4];
  const float* var   = (const float*)d_in[5];
  const float* Wq    = (const float*)d_in[6];
  const float* Wk    = (const float*)d_in[7];
  float* out = (float*)d_out;

  char* ws = (char*)d_ws;
  size_t off = 0;
  auto alloc = [&](size_t bytes) -> void* {
    off = (off + 255) & ~(size_t)255;
    void* p = ws + off;
    off += bytes;
    return p;
  };
  ushort_t* w1c  = (ushort_t*)alloc((size_t)CIN * CH * 2);
  ushort_t* A_t  = (ushort_t*)alloc((size_t)CH * CH * 2);
  ushort_t* f_bf = (ushort_t*)alloc((size_t)B_ * NPTS * CH * 2);
  float*    gbuf = (float*)alloc((size_t)B_ * NPTS * CH * 4);
  float*    sq   = (float*)alloc((size_t)B_ * NPTS * 4);
  int*      idx  = (int*)alloc((size_t)B_ * NPTS * KNN * 4);
  float*    sbuf = (float*)alloc((size_t)B_ * NPTS * 4);

  k_convert_w1<<<(CIN * CH / 2) / 256, 256, 0, stream>>>(W1, w1c);
  k_compute_A<<<CH, CH, 0, stream>>>(Wq, Wk, A_t);
  dim3 grid2(NPTS / TM, B_);
  k_gemm1<<<grid2, 256, 0, stream>>>(x, w1c, gamma, beta, mean, var, f_bf, sq);
  k_gemm_g<<<grid2, 256, 0, stream>>>(f_bf, A_t, gbuf);
  k_knn<<<grid2, 256, 0, stream>>>(f_bf, sq, idx);
  k_attn<<<(B_ * NPTS) / 8, 256, 0, stream>>>(f_bf, gbuf, idx, sbuf);
  k_final<<<B_, 256, 0, stream>>>(sbuf, out);
}
